// Model_52518860096367
// MI455X (gfx1250) — matrix operand order repaired and checked by reading, not compiled, not run on hardware
//
#include <hip/hip_runtime.h>
#include <math.h>

typedef __attribute__((ext_vector_type(16))) __bf16 v16bf;
typedef __attribute__((ext_vector_type(2)))  __bf16 bf16x2;
typedef __attribute__((ext_vector_type(8)))  float  v8f;

#define SQ    2048
#define DH    64
#define KT    64          // key tile width
#define WAVES 4           // waves per block, each owns 16 query rows
#define QROWS (16 * WAVES)
#define KS    72          // ldsK row stride (ushort units), padded, 144B (16B-aligned)
#define VS    72          // ldsV row stride
#define PS    40          // ldsP row stride

static __device__ __forceinline__ unsigned short f2bf_bits(float f) {
    __bf16 h = (__bf16)f;
    return __builtin_bit_cast(unsigned short, h);
}

union BFVec { uint4 u[2]; v16bf v; };

// load 16 contiguous bf16 (32B) from LDS as two b128s
static __device__ __forceinline__ v16bf ldv16(const unsigned short* p) {
    BFVec t;
    t.u[0] = *(const uint4*)(p);
    t.u[1] = *(const uint4*)(p + 8);
    return t.v;
}

static __device__ __forceinline__ unsigned int pk2(float a, float b) {
    bf16x2 p; p[0] = (__bf16)a; p[1] = (__bf16)b;   // -> v_cvt_pk_bf16_f32
    return __builtin_bit_cast(unsigned int, p);
}

#define WMMA_BF16(A, B, C) \
    __builtin_amdgcn_wmma_f32_16x16x32_bf16(false, (A), false, (B), (short)0, (C), false, false)

__global__ void __launch_bounds__(128)
fattn_wmma_bf16(const float* __restrict__ q,
                const float* __restrict__ k,   // [B,H,D,S] pre-transposed
                const float* __restrict__ v,
                const int*   __restrict__ scp,
                float* __restrict__ out) {
    // K tile, TRANSPOSED: [t=64][d=64] bf16, stride 72  (B-operand gathers contiguous)
    __shared__ unsigned short ldsK[KT * KS];
    // V tile, TRANSPOSED: [d=64][t=64] bf16, stride 72
    __shared__ unsigned short ldsV[DH * VS];
    // per-wave P staging: [16][64] bf16, stride 40
    __shared__ unsigned short ldsP[WAVES * 16 * PS];

    const int tid  = threadIdx.x;
    const int lane = tid & 31;
    const int wave = tid >> 5;
    const int n    = lane & 15;   // B/C-layout column, A-layout row
    const int hi   = lane >> 4;   // half-wave select

    const int blocks_per_head = SQ / QROWS;                 // 32
    const int head  = blockIdx.x / blocks_per_head;         // 0..31 (b*H+h)
    const int qbase = (blockIdx.x % blocks_per_head) * QROWS + wave * 16;
    const size_t hoff = (size_t)head * SQ * DH;             // same for q/v/out and k (D*S==S*D)
    const float scale = (float)scp[0];

    // ---- Q tile (16 x 64) into A-format bf16, scale folded in ----
    v16bf aq0, aq1;
#pragma unroll
    for (int i = 0; i < 16; ++i) {
        const int kk = i + 16 * hi;                         // K within 32-chunk
        const size_t rowoff = hoff + (size_t)(qbase + n) * DH;
        aq0[i] = (__bf16)(scale * q[rowoff + kk]);
        aq1[i] = (__bf16)(scale * q[rowoff + 32 + kk]);
    }

    // all-ones B matrix: row-sum of P via the matrix unit
    v16bf ones;
#pragma unroll
    for (int i = 0; i < 16; ++i) ones[i] = (__bf16)1.0f;

    v8f accs[4] = {};                                       // 16 x 64 output accum
    float mrow[8], lrow[8];
#pragma unroll
    for (int r = 0; r < 8; ++r) { mrow[r] = -INFINITY; lrow[r] = 0.0f; }

    unsigned short* myP = &ldsP[wave * 16 * PS];

    for (int t0 = 0; t0 < SQ; t0 += KT) {
        __syncthreads();   // previous iteration's LDS consumers done

        // ---- stage K,V -> LDS as bf16, transposing; 1024 4-elem tasks each ----
#pragma unroll
        for (int it = 0; it < 8; ++it) {
            const int task = tid + 128 * it;                // 0..1023
            {   // K: d-quad dq, column t ; global k[d][t] -> ldsK[t][d]
                const int dq = task >> 6;                   // 0..15
                const int t  = task & 63;
                const float* src = &k[hoff + (size_t)(4 * dq) * SQ + t0 + t];
                const unsigned int u0 = pk2(src[0],      src[SQ]);
                const unsigned int u1 = pk2(src[2 * SQ], src[3 * SQ]);
                uint2 uu; uu.x = u0; uu.y = u1;
                *(uint2*)&ldsK[t * KS + 4 * dq] = uu;       // 8B store, aligned
            }
            {   // V: t-quad tq, column d ; global v[t][d] -> ldsV[d][t]
                const int tq = task >> 6;
                const int d  = task & 63;
                const float* src = &v[hoff + (size_t)(t0 + 4 * tq) * DH + d];
                const unsigned int u0 = pk2(src[0],      src[DH]);
                const unsigned int u1 = pk2(src[2 * DH], src[3 * DH]);
                uint2 uu; uu.x = u0; uu.y = u1;
                *(uint2*)&ldsV[d * VS + 4 * tq] = uu;
            }
        }
        __syncthreads();

        // ---- scores: 4 column tiles of 16, each = 2 WMMAs over D ----
        v8f cs[4] = {};
#pragma unroll
        for (int cc = 0; cc < 4; ++cc) {
            const unsigned short* base = &ldsK[(16 * cc + n) * KS + 16 * hi];
            const v16bf kb0 = ldv16(base);                  // d = 16*hi + i
            const v16bf kb1 = ldv16(base + 32);             // d = 32 + 16*hi + i
            cs[cc] = WMMA_BF16(aq0, kb0, cs[cc]);
            cs[cc] = WMMA_BF16(aq1, kb1, cs[cc]);
        }

        // ---- online softmax: row max via shuffle tree, P -> LDS bf16 ----
        float sfac[8];
#pragma unroll
        for (int r = 0; r < 8; ++r) {
            float mx = fmaxf(fmaxf(cs[0][r], cs[1][r]), fmaxf(cs[2][r], cs[3][r]));
            mx = fmaxf(mx, __shfl_xor(mx, 1, 32));
            mx = fmaxf(mx, __shfl_xor(mx, 2, 32));
            mx = fmaxf(mx, __shfl_xor(mx, 4, 32));
            mx = fmaxf(mx, __shfl_xor(mx, 8, 32));
            const float mnew = fmaxf(mrow[r], mx);
            const float s    = __expf(mrow[r] - mnew);
            mrow[r] = mnew;
            sfac[r] = s;
            const int row = r + 8 * hi;
#pragma unroll
            for (int cc = 0; cc < 4; ++cc) {
                const float e = __expf(cs[cc][r] - mnew);
                myP[row * PS + 16 * cc + n] = f2bf_bits(e);
            }
            accs[0][r] *= s; accs[1][r] *= s; accs[2][r] *= s; accs[3][r] *= s;
        }
        __syncthreads();

        // ---- reload P in A-format (two 16x32 tiles), contiguous b128 loads ----
        const unsigned short* prow = &myP[n * PS];
        BFVec pA0, pA1;
        pA0.u[0] = *(const uint4*)(prow + 16 * hi);
        pA0.u[1] = *(const uint4*)(prow + 8 + 16 * hi);
        pA1.u[0] = *(const uint4*)(prow + 32 + 16 * hi);
        pA1.u[1] = *(const uint4*)(prow + 40 + 16 * hi);
        const v16bf pa0 = pA0.v, pa1 = pA1.v;

        // ---- row sums of P via matrix unit: lsum = P @ ones ----
        v8f lsum = {};
        lsum = WMMA_BF16(pa0, ones, lsum);
        lsum = WMMA_BF16(pa1, ones, lsum);
#pragma unroll
        for (int r = 0; r < 8; ++r) lrow[r] = lrow[r] * sfac[r] + lsum[r];

        // ---- PV: 4 d-chunks x 2 t-halves ----
#pragma unroll
        for (int j = 0; j < 4; ++j) {
            const unsigned short* vbase = &ldsV[(16 * j + n) * VS + 16 * hi];
            const v16bf vb0 = ldv16(vbase);                 // t = 16*hi + i
            const v16bf vb1 = ldv16(vbase + 32);            // t = 32 + 16*hi + i
            accs[j] = WMMA_BF16(pa0, vb0, accs[j]);
            accs[j] = WMMA_BF16(pa1, vb1, accs[j]);
        }
    }

    // ---- epilogue: divide by row sum, store fp32 ----
#pragma unroll
    for (int r = 0; r < 8; ++r) {
        const float inv = 1.0f / lrow[r];
        const size_t row = (size_t)(qbase + r + 8 * hi);
        float* o = &out[hoff + row * DH + n];
        o[0]  = accs[0][r] * inv;
        o[16] = accs[1][r] * inv;
        o[32] = accs[2][r] * inv;
        o[48] = accs[3][r] * inv;
    }
}

extern "C" void kernel_launch(void* const* d_in, const int* in_sizes, int n_in,
                              void* d_out, int out_size, void* d_ws, size_t ws_size,
                              hipStream_t stream) {
    const float* q  = (const float*)d_in[0];
    const float* k  = (const float*)d_in[1];
    const float* v  = (const float*)d_in[2];
    const int*   sc = (const int*)d_in[3];
    float* out = (float*)d_out;

    const int BH = 2 * 16;                  // 32 heads
    const int blocks_per_head = SQ / QROWS; // 32
    dim3 grid(BH * blocks_per_head);        // 1024 blocks
    dim3 block(32 * WAVES);                 // 128 threads = 4 waves
    fattn_wmma_bf16<<<grid, block, 0, stream>>>(q, k, v, sc, out);
}
